// Conv2d_84413287236285
// MI455X (gfx1250) — compile-verified, matrix operand order checked by reading, not run on hardware
//
#include <hip/hip_runtime.h>
#include <hip/hip_bf16.h>
#include <math.h>

typedef __attribute__((ext_vector_type(16))) __bf16 v16bf;
typedef __attribute__((ext_vector_type(2)))  __bf16 v2bf;
typedef __attribute__((ext_vector_type(8)))  float  v8f;
typedef __attribute__((address_space(3))) unsigned short lds_us3;

#define Bn    16
#define Cc    128
#define Oo    256
#define Hh    96
#define OHs   92
#define PIX   (OHs*OHs)        /* 8464  */
#define TOTAL (Bn*PIX)         /* 135424 = 1058*128 exactly */
#define CHS   (Hh*Hh)          /* 9216  */
#define KPAD  32

static __device__ __forceinline__ unsigned short f2bf(float f) {
  unsigned int u = __float_as_uint(f);
  u += 0x7FFFu + ((u >> 16) & 1u);   // round-to-nearest-even to bf16
  return (unsigned short)(u >> 16);
}

// Pack two floats to bf16x2 via the native convert path.
static __device__ __forceinline__ unsigned int pack_bf16(float a, float b) {
  v2bf p;
  p.x = (__bf16)a;
  p.y = (__bf16)b;
  return __builtin_bit_cast(unsigned int, p);
}

// 16B global -> LDS async copy (gfx1250 GLOBAL_LOAD_ASYNC_TO_LDS_B128).
static __device__ __forceinline__ void async_g2l_b128(const void* g, void* l) {
  unsigned lds_off = (unsigned)(size_t)(lds_us3*)l;      // offset from wave LDS base
  unsigned long long ga = (unsigned long long)g;
  asm volatile("global_load_async_to_lds_b128 %0, %1, off"
               :: "v"(lds_off), "v"(ga) : "memory");
}

static __device__ __forceinline__ void wait_async0() {
#if __has_builtin(__builtin_amdgcn_s_wait_asynccnt)
  __builtin_amdgcn_s_wait_asynccnt(0);
#else
  asm volatile("s_wait_asynccnt 0" ::: "memory");
#endif
}

// ---------------------------------------------------------------------------
// Pack W -> bf16 [O][C][32] (25 taps + 7 zeros), W^2 likewise; softplus(w_sigma)
__global__ __launch_bounds__(256) void prep_weights(
    const float* __restrict__ W, const float* __restrict__ wsig,
    unsigned short* __restrict__ Wb, unsigned short* __restrict__ W2b,
    float* __restrict__ sp)
{
  int idx = blockIdx.x * blockDim.x + threadIdx.x;   // o*C + c
  if (idx >= Oo * Cc) return;
  int o = idx >> 7;
  const float* src = W + idx * 25;
  unsigned short* d1 = Wb  + idx * KPAD;
  unsigned short* d2 = W2b + idx * KPAD;
#pragma unroll
  for (int j = 0; j < KPAD; ++j) {
    float v = (j < 25) ? src[j] : 0.0f;
    d1[j] = f2bf(v);
    d2[j] = f2bf(v * v);
  }
  if ((idx & 127) == 0) {
    float a = fmaxf(wsig[o], -88.0f);
    sp[o] = logf(1.0f + expf(a));
  }
}

// ---------------------------------------------------------------------------
// tmp[b,h,w] = sum_c (sigma_x + mu_x^2)
__global__ __launch_bounds__(256) void chan_reduce(
    const float* __restrict__ mu, const float* __restrict__ sg,
    float* __restrict__ tmp)
{
  int idx = blockIdx.x * blockDim.x + threadIdx.x;
  if (idx >= Bn * CHS) return;
  int b = idx / CHS;
  int hw = idx - b * CHS;
  const float* pm = mu + (size_t)b * Cc * CHS + hw;
  const float* ps = sg + (size_t)b * Cc * CHS + hw;
  float s = 0.0f;
  for (int c = 0; c < Cc; ++c) {
    float m = pm[c * CHS];
    s = s + ps[c * CHS] + m * m;
  }
  tmp[idx] = s;
}

// ---------------------------------------------------------------------------
// ssum[b,oh,ow] = 5x5 box sum of tmp
__global__ __launch_bounds__(256) void box_sum(
    const float* __restrict__ tmp, float* __restrict__ ssum)
{
  int idx = blockIdx.x * blockDim.x + threadIdx.x;
  if (idx >= TOTAL) return;
  int b = idx / PIX;  int p = idx - b * PIX;
  int oh = p / OHs;   int ow = p - oh * OHs;
  const float* base = tmp + b * CHS + oh * Hh + ow;
  float s = 0.0f;
#pragma unroll
  for (int kh = 0; kh < 5; ++kh)
#pragma unroll
    for (int kw = 0; kw < 5; ++kw)
      s += base[kh * Hh + kw];
  ssum[idx] = s;
}

// ---------------------------------------------------------------------------
// Fused implicit-GEMM conv (mu path + w^2 sigma path) via bf16 WMMA.
// Block: 256 threads (8 waves), tile M=128 pixels x N=128 out-channels, BK=32
// (= one input channel, 25 taps + 7 zero pad).
union FragU { uint4 q[2]; v16bf v; };

__global__ __launch_bounds__(256) void fused_conv_wmma(
    const float* __restrict__ mu, const float* __restrict__ sg,
    const float* __restrict__ bias,
    const unsigned short* __restrict__ Wb, const unsigned short* __restrict__ W2b,
    const float* __restrict__ sp, const float* __restrict__ ssum,
    float* __restrict__ out_mu, float* __restrict__ out_sg)
{
  constexpr int LDR = 40;                       // ushorts per LDS row (80 B, 16B-aligned)
  __shared__ __align__(16) unsigned short sA[2][128 * LDR]; // [0]=mu tile, [1]=sigma tile
  __shared__ __align__(16) unsigned short sB[2][128 * LDR]; // [0]=W tile,  [1]=W^2 tile

  const int tid   = threadIdx.x;
  const int mtile = blockIdx.x;                 // 0..1057 (exact)
  const int ntile = blockIdx.y;                 // 0..1

  // ---- per-thread A-load geometry: each thread owns taps {2q, 2q+1} of 8 rows ----
  const int  q   = tid & 15;
  const int  t0  = 2 * q, t1 = 2 * q + 1;
  const bool ok0 = (t0 < 25), ok1 = (t1 < 25);
  const int  kh0 = t0 / 5, kw0 = t0 - 5 * (t0 / 5);
  const int  kh1 = t1 / 5, kw1 = t1 - 5 * (t1 / 5);
  const int  rowbase = (tid >> 4) * 8;
  int aoff0[8], aoff1[8];
#pragma unroll
  for (int i = 0; i < 8; ++i) {
    int r = mtile * 128 + rowbase + i;          // output pixel index, always < TOTAL
    int b = r / PIX;  int p = r - b * PIX;
    int oh = p / OHs; int ow = p - oh * OHs;
    aoff0[i] = b * (Cc * CHS) + (oh + kh0) * Hh + (ow + kw0);
    aoff1[i] = b * (Cc * CHS) + (oh + kh1) * Hh + (ow + kw1);
  }

  const int wave = tid >> 5, lane = tid & 31;
  const int wm = wave & 3;                      // 4 wave-rows of 32 pixels
  const int wn = wave >> 2;                     // 2 wave-cols of 64 channels
  const int half = lane >> 4, mrow = lane & 15;

  // B staging geometry (async b128 copies, 2 chunks per thread per tensor)
  const int brow0 = tid >> 2,        bq0 = tid & 3;
  const int brow1 = (256 + tid) >> 2, bq1 = (256 + tid) & 3;

  const v8f vzero = {0.f,0.f,0.f,0.f,0.f,0.f,0.f,0.f};
  v8f accM[2][4], accS[2][4];
#pragma unroll
  for (int i = 0; i < 2; ++i)
#pragma unroll
    for (int j = 0; j < 4; ++j) { accM[i][j] = vzero; accS[i][j] = vzero; }

  for (int c = 0; c < Cc; ++c) {
    const int cc = c * CHS;

    // -------- B tiles: async global->LDS b128 (bf16 pre-packed weights) --------
    {
      size_t ge0 = (size_t)((ntile * 128 + brow0) * Cc + c) * KPAD + bq0 * 8;
      size_t ge1 = (size_t)((ntile * 128 + brow1) * Cc + c) * KPAD + bq1 * 8;
      async_g2l_b128(Wb  + ge0, &sB[0][brow0 * LDR + bq0 * 8]);
      async_g2l_b128(W2b + ge0, &sB[1][brow0 * LDR + bq0 * 8]);
      async_g2l_b128(Wb  + ge1, &sB[0][brow1 * LDR + bq1 * 8]);
      async_g2l_b128(W2b + ge1, &sB[1][brow1 * LDR + bq1 * 8]);
    }

    // -------- A tiles (mu, sigma): scalar f32 loads -> bf16x2 packed LDS stores --------
#pragma unroll
    for (int i = 0; i < 8; ++i) {
      int row = rowbase + i;
      float m0 = 0.f, m1 = 0.f, s0 = 0.f, s1 = 0.f;
      if (ok0) { m0 = mu[aoff0[i] + cc]; s0 = sg[aoff0[i] + cc]; }
      if (ok1) { m1 = mu[aoff1[i] + cc]; s1 = sg[aoff1[i] + cc]; }
      *(unsigned int*)&sA[0][row * LDR + t0] = pack_bf16(m0, m1);
      *(unsigned int*)&sA[1][row * LDR + t0] = pack_bf16(s0, s1);
    }
    if (ok0 && (c + 1 < Cc)) {                  // pull next channel toward caches
      __builtin_prefetch((const void*)(mu + aoff0[0] + (c + 1) * CHS), 0, 1);
      __builtin_prefetch((const void*)(sg + aoff0[0] + (c + 1) * CHS), 0, 1);
    }

    wait_async0();                              // our LDS writes landed
    __syncthreads();                            // everyone's writes visible

    // -------- fragments + WMMA --------
    // A 16x32 bf16 lane layout: lanes 0-15 row m, K {0..7,16..23}; lanes 16-31 K {8..15,24..31}
    FragU aM[2], aS[2];
#pragma unroll
    for (int sm = 0; sm < 2; ++sm) {
      const uint4* pm = (const uint4*)&sA[0][(wm * 32 + sm * 16 + mrow) * LDR];
      aM[sm].q[0] = pm[half]; aM[sm].q[1] = pm[2 + half];
      const uint4* ps = (const uint4*)&sA[1][(wm * 32 + sm * 16 + mrow) * LDR];
      aS[sm].q[0] = ps[half]; aS[sm].q[1] = ps[2 + half];
    }
#pragma unroll
    for (int sn = 0; sn < 4; ++sn) {
      FragU bW, b2;
      const uint4* pw = (const uint4*)&sB[0][(wn * 64 + sn * 16 + mrow) * LDR];
      bW.q[0] = pw[half]; bW.q[1] = pw[2 + half];
      const uint4* p2 = (const uint4*)&sB[1][(wn * 64 + sn * 16 + mrow) * LDR];
      b2.q[0] = p2[half]; b2.q[1] = p2[2 + half];
#pragma unroll
      for (int sm = 0; sm < 2; ++sm) {
        accM[sm][sn] = __builtin_amdgcn_wmma_f32_16x16x32_bf16(
            false, aM[sm].v, false, bW.v, (short)0, accM[sm][sn], false, false);
        accS[sm][sn] = __builtin_amdgcn_wmma_f32_16x16x32_bf16(
            false, aS[sm].v, false, b2.v, (short)0, accS[sm][sn], false, false);
      }
    }
    __syncthreads();
  }

  // -------- fused epilogue --------
  // C/D layout: VGPR e, lanes 0-15 -> (M=e, N=lane); lanes 16-31 -> (M=e+8, N=lane-16)
#pragma unroll
  for (int sm = 0; sm < 2; ++sm) {
#pragma unroll
    for (int e = 0; e < 8; ++e) {
      int r = mtile * 128 + wm * 32 + sm * 16 + e + 8 * half;
      int b = r / PIX;  int p = r - b * PIX;
      float ss = ssum[r];
#pragma unroll
      for (int sn = 0; sn < 4; ++sn) {
        int o = ntile * 128 + wn * 64 + sn * 16 + mrow;
        size_t oidx = (size_t)(b * Oo + o) * PIX + p;
        out_mu[oidx] = accM[sm][sn][e] + bias[o];
        out_sg[oidx] = (sp[o] * ss + accS[sm][sn][e]) * 0.001f;
      }
    }
  }
}

// ---------------------------------------------------------------------------
extern "C" void kernel_launch(void* const* d_in, const int* in_sizes, int n_in,
                              void* d_out, int out_size, void* d_ws, size_t ws_size,
                              hipStream_t stream) {
  const float* mu   = (const float*)d_in[0];
  const float* sgx  = (const float*)d_in[1];
  const float* W    = (const float*)d_in[2];
  const float* bias = (const float*)d_in[3];
  const float* wsig = (const float*)d_in[4];

  char* ws = (char*)d_ws;
  unsigned short* Wb  = (unsigned short*)(ws + 0);          // 2 MB
  unsigned short* W2b = (unsigned short*)(ws + 2097152);    // 2 MB
  float* sp   = (float*)(ws + 4194304);                     // 1 KB
  float* ssum = (float*)(ws + 4195328);                     // 529 KB
  float* tmp  = (float*)(ws + 4737024);                     // 576 KB

  float* out_mu = (float*)d_out;
  float* out_sg = out_mu + (size_t)Bn * Oo * PIX;

  prep_weights<<<(Oo * Cc + 255) / 256, 256, 0, stream>>>(W, wsig, Wb, W2b, sp);
  chan_reduce<<<(Bn * CHS + 255) / 256, 256, 0, stream>>>(mu, sgx, tmp);
  box_sum<<<(TOTAL + 255) / 256, 256, 0, stream>>>(tmp, ssum);

  dim3 grid(TOTAL / 128, Oo / 128);   // (1058, 2)
  fused_conv_wmma<<<grid, 256, 0, stream>>>(mu, sgx, bias, Wb, W2b, sp, ssum,
                                            out_mu, out_sg);
}
